// CRF_90417651516568
// MI455X (gfx1250) — matrix operand order repaired and checked by reading, not compiled, not run on hardware
//
#include <hip/hip_runtime.h>
#include <hip/hip_bf16.h>

typedef __attribute__((ext_vector_type(16))) __bf16 v16bf;
typedef __attribute__((ext_vector_type(8)))  __bf16 v8bf;
typedef __attribute__((ext_vector_type(8)))  float  v8f;

#define T_DIM   512
#define N_BATCH 16
#define D_DIM   512
#define NC      4096            // GEMM output columns (pre-blank)
#define COUT    5120            // with blank columns interleaved
#define M_DIM   (N_BATCH * T_DIM) // 8192 rows (n-major => directly (N,T,C) layout)
#define NSTATES 1024
#define LMAX    256
#define LN      252             // Lmax - (STATE_LEN-1)
#define NEGF    (-1e30f)
#define COPY_OFF (16LL * 512LL * 5120LL)

#define BM 128
#define BN 64
#define BK 32
#define NCHUNK (D_DIM / BK)     // 16
#define LDA 40
#define LDB 40

// branch-free fast tanh: one v_exp_f32, no EXEC-mask divergence
__device__ __forceinline__ float fast_tanh(float x) {
  const float ax = fabsf(x);
  const float e  = __expf(-2.0f * ax);
  const float t  = __fdividef(1.0f - e, 1.0f + e);
  return copysignf(t, x);
}

// ---------------------------------------------------------------------------
// Kernel 1: scores = tanh(hs @ W^T + b) * 5, blank columns = 2.0,
// written twice to d_out in (N, T, COUT) layout.
// bf16 hi/lo split (3 WMMAs: hi*hi + hi*lo + lo*hi ~ fp32 accuracy).
// Double-buffered LDS pipeline: global loads for chunk k+1 issued before
// chunk k's WMMAs, so s_wait_loadcnt lands after the matrix work.
// ---------------------------------------------------------------------------
__global__ __launch_bounds__(256) void crf_gemm_tanh_kernel(
    const float* __restrict__ hs,    // (8192, 512) row-major == hs_pad flat
    const float* __restrict__ Wm,    // (4096, 512)
    const float* __restrict__ bias,  // (4096,)
    float* __restrict__ out)         // scores base (first copy)
{
  __shared__ __bf16 Ah[2][BM * LDA];
  __shared__ __bf16 Al[2][BM * LDA];
  __shared__ __bf16 Bh[2][BN * LDB];
  __shared__ __bf16 Bl[2][BN * LDB];

  const int tid  = threadIdx.x;
  const int lane = tid & 31;
  const int wave = tid >> 5;
  const int wm   = wave >> 1;     // 0..3  (32-row strips)
  const int wn   = wave & 1;      // 0..1  (32-col strips)
  const long bm  = (long)blockIdx.y * BM;
  const long cb  = (long)blockIdx.x * BN;

  v8f acc[2][2] = {};

  const int halfsel = lane >> 4;   // 0 or 1
  const int mr      = lane & 15;
  const int aktop   = halfsel * 8;
  const int bktop   = halfsel * 8;

  // staging assignments
  const int aRow = tid >> 1, aHalf = tid & 1;   // A: 16 floats each
  const int bCol = tid >> 2, bQ    = tid & 3;   // B: 8 floats each
  const float* aPtr = hs + (bm + aRow) * D_DIM + aHalf * 16;
  const float* bPtr = Wm + (cb + bCol) * D_DIM + bQ * 8;

  float4 aP[4];
  float4 bP[2];

  // ---- prologue: load + stage chunk 0 into buffer 0
  #pragma unroll
  for (int u = 0; u < 4; ++u) aP[u] = *(const float4*)(aPtr + u * 4);
  bP[0] = *(const float4*)(bPtr + 0);
  bP[1] = *(const float4*)(bPtr + 4);
  {
    __bf16* dh = &Ah[0][aRow * LDA + aHalf * 16];
    __bf16* dl = &Al[0][aRow * LDA + aHalf * 16];
    const float* af = (const float*)aP;
    #pragma unroll
    for (int e = 0; e < 16; ++e) {
      const float x  = af[e];
      const __bf16 hi = (__bf16)x;
      dh[e] = hi;
      dl[e] = (__bf16)(x - (float)hi);
    }
    __bf16* eh = &Bh[0][bCol * LDB + bQ * 8];
    __bf16* el = &Bl[0][bCol * LDB + bQ * 8];
    const float* bf = (const float*)bP;
    #pragma unroll
    for (int e = 0; e < 8; ++e) {
      const float x  = bf[e];
      const __bf16 hi = (__bf16)x;
      eh[e] = hi;
      el[e] = (__bf16)(x - (float)hi);
    }
  }
  __syncthreads();

  for (int chunk = 0; chunk < NCHUNK; ++chunk) {
    const int cur = chunk & 1;
    const int nxt = cur ^ 1;

    // ---- prefetch next chunk from global into registers (wait hides behind WMMAs)
    if (chunk + 1 < NCHUNK) {
      const float* ap = aPtr + (chunk + 1) * BK;
      const float* bp = bPtr + (chunk + 1) * BK;
      #pragma unroll
      for (int u = 0; u < 4; ++u) aP[u] = *(const float4*)(ap + u * 4);
      bP[0] = *(const float4*)(bp + 0);
      bP[1] = *(const float4*)(bp + 4);
    }

    // ---- fragments from current buffer (documented CDNA5 bf16 layouts)
    v16bf afh[2], afl[2], bfh[2], bfl[2];
    #pragma unroll
    for (int i = 0; i < 2; ++i) {
      const int row = wm * 32 + i * 16 + mr;
      const __bf16* ph = &Ah[cur][row * LDA];
      const __bf16* pl = &Al[cur][row * LDA];
      v8bf r0h = *(const v8bf*)(ph + aktop);
      v8bf r1h = *(const v8bf*)(ph + 16 + aktop);
      v8bf r0l = *(const v8bf*)(pl + aktop);
      v8bf r1l = *(const v8bf*)(pl + 16 + aktop);
      #pragma unroll
      for (int e = 0; e < 8; ++e) {
        afh[i][e] = r0h[e]; afh[i][e + 8] = r1h[e];
        afl[i][e] = r0l[e]; afl[i][e + 8] = r1l[e];
      }
    }
    #pragma unroll
    for (int j = 0; j < 2; ++j) {
      const int col = wn * 32 + j * 16 + mr;
      const __bf16* ph = &Bh[cur][col * LDB + bktop];
      const __bf16* pl = &Bl[cur][col * LDB + bktop];
      v8bf c0h = *(const v8bf*)(ph);
      v8bf c1h = *(const v8bf*)(ph + 16);
      v8bf c0l = *(const v8bf*)(pl);
      v8bf c1l = *(const v8bf*)(pl + 16);
      #pragma unroll
      for (int e = 0; e < 8; ++e) {
        bfh[j][e] = c0h[e]; bfh[j][e + 8] = c1h[e];
        bfl[j][e] = c0l[e]; bfl[j][e + 8] = c1l[e];
      }
    }

    #pragma unroll
    for (int i = 0; i < 2; ++i)
      #pragma unroll
      for (int j = 0; j < 2; ++j) {
        acc[i][j] = __builtin_amdgcn_wmma_f32_16x16x32_bf16(
            false, afh[i], false, bfh[j], (short)0, acc[i][j], false, false);
        acc[i][j] = __builtin_amdgcn_wmma_f32_16x16x32_bf16(
            false, afh[i], false, bfl[j], (short)0, acc[i][j], false, false);
        acc[i][j] = __builtin_amdgcn_wmma_f32_16x16x32_bf16(
            false, afl[i], false, bfh[j], (short)0, acc[i][j], false, false);
      }

    __syncthreads();

    // ---- stage next chunk into alternate buffer
    if (chunk + 1 < NCHUNK) {
      __bf16* dh = &Ah[nxt][aRow * LDA + aHalf * 16];
      __bf16* dl = &Al[nxt][aRow * LDA + aHalf * 16];
      const float* af = (const float*)aP;
      #pragma unroll
      for (int e = 0; e < 16; ++e) {
        const float x  = af[e];
        const __bf16 hi = (__bf16)x;
        dh[e] = hi;
        dl[e] = (__bf16)(x - (float)hi);
      }
      __bf16* eh = &Bh[nxt][bCol * LDB + bQ * 8];
      __bf16* el = &Bl[nxt][bCol * LDB + bQ * 8];
      const float* bf = (const float*)bP;
      #pragma unroll
      for (int e = 0; e < 8; ++e) {
        const float x  = bf[e];
        const __bf16 hi = (__bf16)x;
        eh[e] = hi;
        el[e] = (__bf16)(x - (float)hi);
      }
    }
    __syncthreads();
  }

  // Epilogue: D layout: col = lane&15, rows = (lane>=16)*8 + r
  const int rtop = (lane >> 4) * 8;
  #pragma unroll
  for (int i = 0; i < 2; ++i) {
    #pragma unroll
    for (int j = 0; j < 2; ++j) {
      const long c  = cb + wn * 32 + j * 16 + mr;
      const float bv = bias[c];
      const long oc = (c >> 2) * 5 + 1 + (c & 3);
      #pragma unroll
      for (int r = 0; r < 8; ++r) {
        const long row = bm + wm * 32 + i * 16 + rtop + r;
        const float v  = fast_tanh(acc[i][j][r] + bv) * 5.0f;
        out[row * COUT + oc]            = v;
        out[COPY_OFF + row * COUT + oc] = v;
      }
    }
  }
  // blank columns (value 2.0): 16 groups x 128 rows for this block
  for (int idx = tid; idx < 16 * BM; idx += 256) {
    const int  g   = idx & 15;
    const long row = bm + (idx >> 4);
    const long oc  = ((cb >> 2) + g) * 5;
    out[row * COUT + oc]            = 2.0f;
    out[COPY_OFF + row * COUT + oc] = 2.0f;
  }
}

// ---------------------------------------------------------------------------
// Kernel 2: full-state forward scan.  One block per batch element.
// alpha ping-pong in LDS, 1 barrier per timestep.  logZ[n] -> d_ws.
// IDX[s] = [s, s/4, 256+s/4, 512+s/4, 768+s/4]; M columns j=0..4 interleaved.
// global_prefetch of next timestep row overlaps the barrier+transcendentals.
// ---------------------------------------------------------------------------
__global__ __launch_bounds__(1024) void crf_logz_kernel(
    const float* __restrict__ scores,  // (N, T, COUT)
    float* __restrict__ logZ)          // (N,)
{
  __shared__ float aA[NSTATES];
  __shared__ float aB[NSTATES];
  const int s = threadIdx.x;
  const int n = blockIdx.x;
  const int p = s >> 2;
  aA[s] = 0.0f;
  __syncthreads();

  float* cur = aA;
  float* nxt = aB;
  const float* base = scores + (long)n * T_DIM * COUT;
  for (int t = 0; t < T_DIM; ++t) {
    const float* M = base + (long)t * COUT + 5 * s;
    __builtin_prefetch(M + COUT, 0, 0);   // next timestep row -> global_prefetch_b8
    const float v0 = cur[s]       + M[0];
    const float v1 = cur[p]       + M[1];
    const float v2 = cur[256 + p] + M[2];
    const float v3 = cur[512 + p] + M[3];
    const float v4 = cur[768 + p] + M[4];
    const float m  = fmaxf(fmaxf(fmaxf(v0, v1), fmaxf(v2, v3)), v4);
    const float sum = __expf(v0 - m) + __expf(v1 - m) + __expf(v2 - m) +
                      __expf(v3 - m) + __expf(v4 - m);
    nxt[s] = m + __logf(sum);
    __syncthreads();
    float* tmp = cur; cur = nxt; nxt = tmp;
  }
  if (s == 0) {
    float m = cur[0];
    for (int i = 1; i < NSTATES; ++i) m = fmaxf(m, cur[i]);
    float sum = 0.0f;
    for (int i = 0; i < NSTATES; ++i) sum += __expf(cur[i] - m);
    logZ[n] = m + __logf(sum);
  }
}

// ---------------------------------------------------------------------------
// Kernel 3: CTC forward on RAW scores; subtract logZ[n] at the end
// (norm = scores - logZ/T accumulates exactly logZ over T steps).
// ---------------------------------------------------------------------------
__global__ __launch_bounds__(256) void crf_ctc_kernel(
    const float* __restrict__ scores,   // (N, T, COUT)
    const int*   __restrict__ ys_pad,   // (N, 256)
    const int*   __restrict__ ys_lens,  // (N,)
    const float* __restrict__ logZ,     // (N,)
    float* __restrict__ loss)           // (N,)
{
  __shared__ int   tb[LMAX];
  __shared__ int   stayi[LN];
  __shared__ int   movei[LN - 1];
  __shared__ float aA[LN];
  __shared__ float aB[LN];
  const int l = threadIdx.x;
  const int n = blockIdx.x;

  tb[l] = max(ys_pad[n * LMAX + l] - 1, 0);
  __syncthreads();
  if (l < LN) {
    const int kmer = tb[l] * 256 + tb[l + 1] * 64 + tb[l + 2] * 16 +
                     tb[l + 3] * 4 + tb[l + 4];
    stayi[l] = kmer * 5;
    aA[l] = (l == 0) ? 0.0f : NEGF;
  }
  __syncthreads();
  if (l < LN - 1) movei[l] = stayi[l + 1] + tb[l] + 1;
  __syncthreads();

  float* cur = aA;
  float* nxt = aB;
  const float* base = scores + (long)n * T_DIM * COUT;
  for (int t = 0; t < T_DIM; ++t) {
    if (l < LN) {
      const float* row = base + (long)t * COUT;
      __builtin_prefetch(row + COUT + stayi[l], 0, 0);  // next-step gather
      const float a = cur[l] + row[stayi[l]];
      const float b = (l >= 1) ? (cur[l - 1] + row[movei[l - 1]]) : NEGF;
      const float m = fmaxf(a, b);
      const float d = fminf(a, b) - m;
      nxt[l] = m + __logf(1.0f + __expf(d));
    }
    __syncthreads();
    float* tmp = cur; cur = nxt; nxt = tmp;
  }
  if (l == 0) {
    const int   yl   = ys_lens[n];
    const int   lens = yl - 4;               // ys_lens + 1 - STATE_LEN
    const float lz   = cur[lens - 1] - logZ[n];
    loss[n] = -(lz / (float)yl);
  }
}

// ---------------------------------------------------------------------------
extern "C" void kernel_launch(void* const* d_in, const int* in_sizes, int n_in,
                              void* d_out, int out_size, void* d_ws, size_t ws_size,
                              hipStream_t stream) {
  const float* hs      = (const float*)d_in[0];  // hs_pad (16,512,512)
  const int*   ys_pad  = (const int*)  d_in[2];  // (16,256)
  const int*   ys_lens = (const int*)  d_in[3];  // (16,)
  const float* Wm      = (const float*)d_in[4];  // (4096,512)
  const float* bias    = (const float*)d_in[5];  // (4096,)

  float* out     = (float*)d_out;
  float* loss    = out;         // [0, 16)
  float* scores1 = out + 16;    // (16, 512, 5120), second copy at +COPY_OFF
  float* logZ    = (float*)d_ws;

  dim3 gemm_grid(NC / BN, M_DIM / BM);   // 64 x 64
  crf_gemm_tanh_kernel<<<gemm_grid, dim3(256), 0, stream>>>(hs, Wm, bias, scores1);
  crf_logz_kernel<<<dim3(N_BATCH), dim3(NSTATES), 0, stream>>>(scores1, logZ);
  crf_ctc_kernel<<<dim3(N_BATCH), dim3(256), 0, stream>>>(scores1, ys_pad, ys_lens,
                                                          logZ, loss);
}
